// MultiHeadAttention_1142461300950
// MI455X (gfx1250) — matrix operand order repaired and checked by reading, not compiled, not run on hardware
//
#include <hip/hip_runtime.h>
#include <hip/hip_bf16.h>

typedef __attribute__((ext_vector_type(16))) __bf16        v16bf;
typedef __attribute__((ext_vector_type(8)))  float         v8f;
typedef __attribute__((ext_vector_type(8)))  unsigned int  v8u;
typedef __attribute__((ext_vector_type(4)))  unsigned int  v4u;

__device__ __forceinline__ unsigned short f2bfbits(float f) {
    unsigned u = __builtin_bit_cast(unsigned, f);
    u += 0x7fffu + ((u >> 16) & 1u);            // round-to-nearest-even
    return (unsigned short)(u >> 16);
}

// LDS byte offset of a __shared__ object (generic pointer low 32 bits = LDS offset)
template <typename T>
__device__ __forceinline__ unsigned lds_off(T* p) {
    return (unsigned)(reinterpret_cast<uintptr_t>(p));
}

// gfx1250 async copy: 16 bytes global -> LDS, tracked by ASYNCcnt
__device__ __forceinline__ void async_copy_b128(unsigned lds_byte_off, const void* g) {
    asm volatile("global_load_async_to_lds_b128 %0, %1, off"
                 :: "v"(lds_byte_off), "v"((unsigned long long)(uintptr_t)g)
                 : "memory");
}
__device__ __forceinline__ void wait_async0() {
    asm volatile("s_wait_asynccnt 0" ::: "memory");
}

// ---------------------------------------------------------------------------
// f32 -> bf16 pre-conversion (8 elements / thread, vectorized store)
// ---------------------------------------------------------------------------
__global__ __launch_bounds__(256)
void cvt_f32_bf16(const float* __restrict__ in, unsigned short* __restrict__ out, int n)
{
    int i = (blockIdx.x * 256 + threadIdx.x) * 8;
    if (i + 8 <= n) {
        float f[8];
        #pragma unroll
        for (int j = 0; j < 8; ++j) f[j] = in[i + j];
        v4u p;
        #pragma unroll
        for (int j = 0; j < 4; ++j)
            p[j] = (unsigned)f2bfbits(f[2 * j]) | ((unsigned)f2bfbits(f[2 * j + 1]) << 16);
        *(v4u*)(out + i) = p;
    }
}

// ---------------------------------------------------------------------------
// Generic bf16 GEMM: Y[M,1024] = A[M,1024] @ W[1024,1024]^T + bias
// MODE 0: write f32 row-major to outF (final projection)
// MODE 1: write bf16 [B,H,N,D] with RoPE applied (Q, K)
// MODE 2: write bf16 [B,H,N,D] without RoPE (V)
// Block: 128 threads (4 waves). Block tile 64x64, wave tile 16x64.
// Tiles staged via global_load_async_to_lds_b128.
// ---------------------------------------------------------------------------
template <int MODE>
__global__ __launch_bounds__(128)
void gemm_wmma(const unsigned short* __restrict__ A, const unsigned short* __restrict__ W,
               const float* __restrict__ bias,
               float* __restrict__ outF, unsigned short* __restrict__ outBF)
{
    constexpr int K = 1024, NOUT = 1024;
    constexpr int Nseq = 2048, Hh = 16, Dd = 64;

    __shared__ unsigned short a_lds[64][32];     // A tile  (m, k) bf16
    __shared__ unsigned short b_lds[64][32];     // W tile  (n, k) bf16
    __shared__ float          c_lds[4][16][64];  // per-wave result tile (MODE 1/2)

    const int t    = threadIdx.x;
    const int wave = t >> 5, lane = t & 31;
    const int row  = lane & 15, half = lane >> 4;
    const int m0   = blockIdx.x * 64;
    const int nb   = blockIdx.y * 64;

    const int srow = t >> 1, shalf = (t & 1) * 16;   // staging: 16 contiguous bf16/thread
    const unsigned la = lds_off(&a_lds[srow][shalf]);
    const unsigned lb = lds_off(&b_lds[srow][shalf]);

    v8f acc[4] = {};

    for (int kb = 0; kb < K; kb += 32) {
        __syncthreads();                              // previous tile fully consumed
        const unsigned short* ga = A + (size_t)(m0 + srow) * K + kb + shalf;
        const unsigned short* gw = W + (size_t)(nb + srow) * K + kb + shalf;
        async_copy_b128(la,      ga);
        async_copy_b128(la + 16, ga + 8);
        async_copy_b128(lb,      gw);
        async_copy_b128(lb + 16, gw + 8);
        wait_async0();
        __syncthreads();

        // A fragment: 16x32 bf16 (ISA 7.12.2 layout)
        v8u au;
        #pragma unroll
        for (int v = 0; v < 8; ++v) {
            int kk = (v < 4 ? v * 2 : 16 + (v - 4) * 2) + half * 8;
            au[v] = *(const unsigned int*)&a_lds[wave * 16 + row][kk];
        }
        v16bf af = __builtin_bit_cast(v16bf, au);

        #pragma unroll
        for (int c = 0; c < 4; ++c) {
            v8u bu;
            #pragma unroll
            for (int v = 0; v < 8; ++v) {
                int kk = (v < 4 ? v * 2 : 16 + (v - 4) * 2) + half * 8;
                bu[v] = *(const unsigned int*)&b_lds[c * 16 + row][kk];
            }
            v16bf bf = __builtin_bit_cast(v16bf, bu);
            acc[c] = __builtin_amdgcn_wmma_f32_16x16x32_bf16(
                false, af, false, bf, (short)0, acc[c], false, false);
        }
    }

    if (MODE == 0) {
        #pragma unroll
        for (int c = 0; c < 4; ++c) {
            int col  = nb + c * 16 + row;
            float bv = bias[col];
            #pragma unroll
            for (int r = 0; r < 8; ++r) {
                int gm = m0 + wave * 16 + r + 8 * half;
                outF[(size_t)gm * NOUT + col] = acc[c][r] + bv;
            }
        }
    } else {
        // dump wave tile to LDS for cross-column RoPE + bf16 [B,H,N,D] store
        #pragma unroll
        for (int c = 0; c < 4; ++c) {
            int col  = c * 16 + row;
            float bv = bias[nb + col];
            #pragma unroll
            for (int r = 0; r < 8; ++r)
                c_lds[wave][r + 8 * half][col] = acc[c][r] + bv;
        }
        __syncthreads();
        const int h = nb / Dd;                   // output tile == one head
        for (int r = 0; r < 16; ++r) {
            int gm = m0 + wave * 16 + r;
            int b  = gm / Nseq, n = gm % Nseq;
            size_t obase = ((size_t)(b * Hh + h) * Nseq + n) * Dd;
            #pragma unroll
            for (int hf = 0; hf < 2; ++hf) {
                int d   = hf * 32 + lane;
                float y = c_lds[wave][r][d];
                float o;
                if (MODE == 2) {
                    o = y;
                } else {
                    int j = d & 31;              // inv_freq index
                    float th = (float)n * __expf(-0.28782313662425572f * (float)j);
                    float sn, cs;
                    __sincosf(th, &sn, &cs);
                    float y2 = (d < 32) ? c_lds[wave][r][2 * j + 1]
                                        : c_lds[wave][r][2 * j];
                    o = (d < 32) ? (y * cs - y2 * sn) : (y * cs + y2 * sn);
                }
                outBF[obase + d] = f2bfbits(o);
            }
        }
    }
}

// ---------------------------------------------------------------------------
// Flash attention (causal, online softmax). Q,K,V bf16 [B,H,N,D].
// Block: 128 threads (4 waves) handles 64 query rows of one (b,h).
// Wave owns 16 query rows; K/V streamed in 32-wide chunks via LDS
// (K chunk via async-to-LDS; V chunk transposed through registers).
// attn output: bf16 [B,N,H*D].
// ---------------------------------------------------------------------------
__global__ __launch_bounds__(128)
void flash_attn(const unsigned short* __restrict__ Q,
                const unsigned short* __restrict__ Kb,
                const unsigned short* __restrict__ Vb,
                unsigned short* __restrict__ attn)
{
    constexpr int Nseq = 2048, Hh = 16, Dd = 64, Cc = 1024;
    const float scale = 0.125f;                  // 1/sqrt(64)

    __shared__ unsigned short k_lds[32][64];     // K chunk (krow, d)
    __shared__ unsigned short v_lds[64][36];     // V chunk transposed (d, krow), padded
    __shared__ float          s_lds[4][16][32];  // per-wave scores
    __shared__ unsigned short p_lds[4][16][32];  // per-wave probs (bf16)
    __shared__ float          alpha_lds[4][16];
    __shared__ float          l_lds[4][16];

    const int t = threadIdx.x, wave = t >> 5, lane = t & 31;
    const int row = lane & 15, half = lane >> 4;
    const int qt = blockIdx.x, h = blockIdx.y, b = blockIdx.z;
    const int qr0 = qt * 64 + wave * 16;
    const size_t headbase = ((size_t)(b * Hh + h)) * Nseq * Dd;

    // Preload Q as two A-fragments (D=64 -> two K=32 steps)
    const unsigned short* qp = Q + headbase + (size_t)qr0 * Dd;
    v16bf aQ[2];
    #pragma unroll
    for (int s = 0; s < 2; ++s) {
        v8u u;
        #pragma unroll
        for (int v = 0; v < 8; ++v) {
            int dk = s * 32 + (v < 4 ? v * 2 : 16 + (v - 4) * 2) + half * 8;
            u[v] = *(const unsigned int*)(qp + (size_t)row * Dd + dk);
        }
        aQ[s] = __builtin_bit_cast(v16bf, u);
    }

    v8f o[4] = {};
    float m = -1e30f, l = 0.0f;

    const int skk = t >> 2, sd0 = (t & 3) * 16;  // staging: 16 contiguous bf16/thread
    const unsigned lk = lds_off(&k_lds[skk][sd0]);

    const int kend = qt * 64 + 64;               // causal horizon for this block
    for (int kc = 0; kc < kend; kc += 32) {
        __syncthreads();
        const unsigned short* kp = Kb + headbase + (size_t)kc * Dd;
        const unsigned short* vp = Vb + headbase + (size_t)kc * Dd;
        // K chunk: async copy straight to LDS
        async_copy_b128(lk,      kp + (size_t)skk * Dd + sd0);
        async_copy_b128(lk + 16, kp + (size_t)skk * Dd + sd0 + 8);
        // V chunk: vectorized global read, transposed 16-bit LDS stores
        v4u a0 = *(const v4u*)(vp + (size_t)skk * Dd + sd0);
        v4u a1 = *(const v4u*)(vp + (size_t)skk * Dd + sd0 + 8);
        #pragma unroll
        for (int j = 0; j < 4; ++j) {
            v_lds[sd0 + 2 * j    ][skk] = (unsigned short)(a0[j] & 0xffffu);
            v_lds[sd0 + 2 * j + 1][skk] = (unsigned short)(a0[j] >> 16);
            v_lds[sd0 + 8 + 2 * j    ][skk] = (unsigned short)(a1[j] & 0xffffu);
            v_lds[sd0 + 8 + 2 * j + 1][skk] = (unsigned short)(a1[j] >> 16);
        }
        wait_async0();
        __syncthreads();

        // S(16x32) = Q(16x64) * K^T(64x32): two col-halves x two K-steps
        #pragma unroll
        for (int c = 0; c < 2; ++c) {
            v8f sc = {};
            #pragma unroll
            for (int s = 0; s < 2; ++s) {
                v8u u;
                #pragma unroll
                for (int v = 0; v < 8; ++v) {
                    int dk = s * 32 + (v < 4 ? v * 2 : 16 + (v - 4) * 2) + half * 8;
                    u[v] = *(const unsigned int*)&k_lds[c * 16 + row][dk];
                }
                v16bf bf = __builtin_bit_cast(v16bf, u);
                sc = __builtin_amdgcn_wmma_f32_16x16x32_bf16(
                    false, aQ[s], false, bf, (short)0, sc, false, false);
            }
            #pragma unroll
            for (int r = 0; r < 8; ++r) {
                int rr = r + 8 * half;
                int kcol = kc + c * 16 + row;
                s_lds[wave][rr][c * 16 + row] =
                    (kcol <= qr0 + rr) ? sc[r] * scale : -1e30f;
            }
        }
        __syncthreads();

        // Online softmax: lanes 0..15, one query row each
        if (lane < 16) {
            float vals[32];
            float cmax = -1e30f;
            #pragma unroll
            for (int cI = 0; cI < 32; ++cI) {
                vals[cI] = s_lds[wave][lane][cI];
                cmax = fmaxf(cmax, vals[cI]);
            }
            float mnew  = fmaxf(m, cmax);
            float alpha = __expf(m - mnew);
            float rs = 0.f;
            #pragma unroll
            for (int cI = 0; cI < 32; ++cI) {
                float p = __expf(vals[cI] - mnew);
                rs += p;
                p_lds[wave][lane][cI] = f2bfbits(p);
            }
            l = l * alpha + rs;
            m = mnew;
            alpha_lds[wave][lane] = alpha;
        }
        __syncthreads();

        // Rescale O, then O(16x64) += P(16x32) * V(32x64)
        float ar[8];
        #pragma unroll
        for (int r = 0; r < 8; ++r) ar[r] = alpha_lds[wave][r + 8 * half];
        #pragma unroll
        for (int dg = 0; dg < 4; ++dg)
            #pragma unroll
            for (int r = 0; r < 8; ++r) o[dg][r] *= ar[r];

        v8u pu;
        #pragma unroll
        for (int v = 0; v < 8; ++v) {
            int kk = (v < 4 ? v * 2 : 16 + (v - 4) * 2) + half * 8;
            pu[v] = *(const unsigned int*)&p_lds[wave][row][kk];
        }
        v16bf pf = __builtin_bit_cast(v16bf, pu);
        #pragma unroll
        for (int dg = 0; dg < 4; ++dg) {
            v8u vu;
            #pragma unroll
            for (int v = 0; v < 8; ++v) {
                int kk = (v < 4 ? v * 2 : 16 + (v - 4) * 2) + half * 8;
                vu[v] = *(const unsigned int*)&v_lds[dg * 16 + row][kk];
            }
            v16bf vf = __builtin_bit_cast(v16bf, vu);
            o[dg] = __builtin_amdgcn_wmma_f32_16x16x32_bf16(
                false, pf, false, vf, (short)0, o[dg], false, false);
        }
    }

    if (lane < 16) l_lds[wave][lane] = l;
    __syncthreads();
    float linv[8];
    #pragma unroll
    for (int r = 0; r < 8; ++r) {
        float lv = l_lds[wave][r + 8 * half];
        linv[r] = lv > 0.f ? 1.0f / lv : 0.f;
    }
    #pragma unroll
    for (int dg = 0; dg < 4; ++dg) {
        int col = h * 64 + dg * 16 + row;
        #pragma unroll
        for (int r = 0; r < 8; ++r) {
            int gq = qr0 + r + 8 * half;
            attn[((size_t)(b * Nseq + gq)) * Cc + col] = f2bfbits(o[dg][r] * linv[r]);
        }
    }
}

// ---------------------------------------------------------------------------
extern "C" void kernel_launch(void* const* d_in, const int* in_sizes, int n_in,
                              void* d_out, int out_size, void* d_ws, size_t ws_size,
                              hipStream_t stream)
{
    (void)in_sizes; (void)n_in; (void)out_size; (void)ws_size;
    const float* x  = (const float*)d_in[0];
    const float* Wq = (const float*)d_in[1];
    const float* bq = (const float*)d_in[2];
    const float* Wk = (const float*)d_in[3];
    const float* bk = (const float*)d_in[4];
    const float* Wv = (const float*)d_in[5];
    const float* bv = (const float*)d_in[6];
    const float* Wo = (const float*)d_in[7];
    const float* bo = (const float*)d_in[8];
    float* out = (float*)d_out;

    constexpr int Bb = 2, Nseq = 2048, C = 1024, K = 1024;
    const size_t M = (size_t)Bb * Nseq;          // 4096

    unsigned short* p   = (unsigned short*)d_ws;
    unsigned short* xbf  = p; p += M * K;        // 8 MB
    unsigned short* wqbf = p; p += (size_t)C * K;
    unsigned short* wkbf = p; p += (size_t)C * K;
    unsigned short* wvbf = p; p += (size_t)C * K;
    unsigned short* wobf = p; p += (size_t)C * K;
    unsigned short* Qbf  = p; p += M * C;
    unsigned short* Kbf  = p; p += M * C;
    unsigned short* Vbf  = p; p += M * C;
    unsigned short* attn = p; p += M * C;

    // pre-convert fp32 operands to bf16 (once)
    cvt_f32_bf16<<<(int)(M * K / 2048), 256, 0, stream>>>(x,  xbf,  (int)(M * K));
    cvt_f32_bf16<<<C * K / 2048,        256, 0, stream>>>(Wq, wqbf, C * K);
    cvt_f32_bf16<<<C * K / 2048,        256, 0, stream>>>(Wk, wkbf, C * K);
    cvt_f32_bf16<<<C * K / 2048,        256, 0, stream>>>(Wv, wvbf, C * K);
    cvt_f32_bf16<<<C * K / 2048,        256, 0, stream>>>(Wo, wobf, C * K);

    dim3 blk(128);
    dim3 ggrid((unsigned)(M / 64), C / 64);

    gemm_wmma<1><<<ggrid, blk, 0, stream>>>(xbf, wqbf, bq, nullptr, Qbf);
    gemm_wmma<1><<<ggrid, blk, 0, stream>>>(xbf, wkbf, bk, nullptr, Kbf);
    gemm_wmma<2><<<ggrid, blk, 0, stream>>>(xbf, wvbf, bv, nullptr, Vbf);

    flash_attn<<<dim3(Nseq / 64, 16, Bb), blk, 0, stream>>>(Qbf, Kbf, Vbf, attn);

    gemm_wmma<0><<<ggrid, blk, 0, stream>>>(attn, wobf, bo, out, nullptr);
}
